// E2EBridge_88218628259858
// MI455X (gfx1250) — matrix operand order repaired and checked by reading, not compiled, not run on hardware
//
#include <hip/hip_runtime.h>
#include <hip/hip_bf16.h>
#include <stdint.h>

// ---------------- problem constants (from reference) ----------------
#define POOLED 7
#define CCH    256          // channels
#define RPL    512          // rois per level (B*N = 4*128)
#define NROI   1536         // 3 levels * 512
#define FLATK  12544        // 256 * 7 * 7
#define REP    1024
#define NPB    128          // rois per batch image

// d_out float-element offsets (tuple concatenated flat, return order)
#define O_IDX   0LL
#define O_DEX   1536LL
#define O_VERTS 3072LL          // 1536*778*3 = 3585024
#define O_J3D   3588096LL       // 1536*21*3 = 96768
#define O_J2D   3684864LL       // 1536*21*2 = 64512
#define O_SIDES 3749376LL       // 1536
#define O_H     3750912LL       // 1536*1024 = 1572864
#define O_BIDX  5323776LL       // 1536
#define O_LIDX  5325312LL       // 1536
#define O_TOTAL 5326848LL
#define MISC_TOTAL 3753984LL    // O_TOTAL - 1572864 (everything except h)

typedef __attribute__((ext_vector_type(16))) __bf16        v16bf;
typedef __attribute__((ext_vector_type(8)))  float         v8f;
typedef __attribute__((ext_vector_type(4)))  unsigned int  v4u;

union FragBF { v4u u[2]; v16bf v; };   // 32B WMMA A/B fragment per lane

__device__ __forceinline__ unsigned short f2bf(float f) {
  unsigned u = __float_as_uint(f);
  u += 0x7FFFu + ((u >> 16) & 1u);     // round-to-nearest-even
  return (unsigned short)(u >> 16);
}

// ---------------- 1) ROI-align -> bf16 activation matrix A[1536][12544] ----
// one block per ROI, one thread per channel; bf16 row staged through LDS so
// the final global write is fully coalesced b32 stores.
__global__ __launch_bounds__(256) void roi_align_bf16(
    const float* __restrict__ f0, const float* __restrict__ f1,
    const float* __restrict__ f2,
    const float* __restrict__ bx0, const float* __restrict__ bx1,
    const float* __restrict__ bx2,
    unsigned short* __restrict__ A)
{
  __shared__ unsigned short tile[FLATK];   // 25 KB of the 320 KB WGP LDS
  const int roi = blockIdx.x;
  const int lvl = roi / RPL;
  const int j   = roi - lvl * RPL;
  const int b   = j / NPB;

  const float* feat; const float* bxs; int H; float scale;
  if (lvl == 0)      { feat = f0; bxs = bx0; H = 100; scale = 0.125f;   }
  else if (lvl == 1) { feat = f1; bxs = bx1; H = 50;  scale = 0.0625f;  }
  else               { feat = f2; bxs = bx2; H = 25;  scale = 0.03125f; }
  const int W = H, HW = H * W;

  const float x1 = bxs[j * 4 + 0] * scale;
  const float y1 = bxs[j * 4 + 1] * scale;
  const float x2 = bxs[j * 4 + 2] * scale;
  const float y2 = bxs[j * 4 + 3] * scale;
  const float bw = fmaxf(x2 - x1, 1.f) * (1.f / POOLED);
  const float bh = fmaxf(y2 - y1, 1.f) * (1.f / POOLED);

  const int c = threadIdx.x;
  const float* __restrict__ fp = feat + ((size_t)b * CCH + c) * HW;

  for (int py = 0; py < POOLED; ++py) {
    for (int px = 0; px < POOLED; ++px) {
      float acc = 0.f;
#pragma unroll
      for (int sy = 0; sy < 2; ++sy) {
        float y  = y1 + ((float)py + (sy ? 0.75f : 0.25f)) * bh;
        bool  vy = (y > -1.f) && (y < (float)H);
        float yc = fminf(fmaxf(y, 0.f), (float)(H - 1));
        int   iy0 = (int)floorf(yc);
        int   iy1 = (iy0 + 1 < H - 1) ? iy0 + 1 : H - 1;
        float ly = yc - (float)iy0, hy = 1.f - ly;
#pragma unroll
        for (int sx = 0; sx < 2; ++sx) {
          float x  = x1 + ((float)px + (sx ? 0.75f : 0.25f)) * bw;
          bool  vx = (x > -1.f) && (x < (float)W);
          float xc = fminf(fmaxf(x, 0.f), (float)(W - 1));
          int   ix0 = (int)floorf(xc);
          int   ix1 = (ix0 + 1 < W - 1) ? ix0 + 1 : W - 1;
          float lx = xc - (float)ix0, hx = 1.f - lx;
          if (vy && vx) {
            float v00 = fp[iy0 * W + ix0], v01 = fp[iy0 * W + ix1];
            float v10 = fp[iy1 * W + ix0], v11 = fp[iy1 * W + ix1];
            acc += hy * hx * v00 + hy * lx * v01 + ly * hx * v10 + ly * lx * v11;
          }
        }
      }
      tile[c * 49 + py * POOLED + px] = f2bf(acc * 0.25f);
    }
  }
  __syncthreads();
  const unsigned int* __restrict__ s = (const unsigned int*)tile;
  unsigned int* __restrict__ d = (unsigned int*)(A + (size_t)roi * FLATK);
  for (int i = threadIdx.x; i < FLATK / 2; i += 256) d[i] = s[i];
}

// ---------------- 2) transpose + fp32->bf16 convert: W[R][C] -> Wt[C][R] ----
__global__ __launch_bounds__(256) void transpose_cvt_bf16(
    const float* __restrict__ src, unsigned short* __restrict__ dst,
    int R, int Ccols)
{
  __shared__ float t[32][33];
  const int cb = blockIdx.x * 32, rb = blockIdx.y * 32;
#pragma unroll
  for (int k = 0; k < 4; ++k) {
    int r = rb + threadIdx.y + k * 8;
    t[threadIdx.y + k * 8][threadIdx.x] = src[(size_t)r * Ccols + cb + threadIdx.x];
  }
  __syncthreads();
#pragma unroll
  for (int k = 0; k < 4; ++k) {
    int cc = cb + threadIdx.y + k * 8;
    dst[(size_t)cc * R + rb + threadIdx.x] = f2bf(t[threadIdx.x][threadIdx.y + k * 8]);
  }
}

// ---------------- 3) bf16 WMMA GEMM: relu(A[M][K] * Bt[N][K]^T + bias) ------
// block = 8 waves = 64(M) x 128(N) tile; each wave owns a 32x32 (2x2 of 16x16)
// accumulator block -> 4x v_wmma_f32_16x16x32_bf16 per K-step of 32.
__device__ __forceinline__ void store_tile(v8f c, int rowBase, int col, float bs,
                                           int N, float* __restrict__ outF,
                                           unsigned short* __restrict__ outB)
{
#pragma unroll
  for (int r = 0; r < 8; ++r) {
    float v = c[r] + bs;
    v = v > 0.f ? v : 0.f;                 // ReLU
    size_t idx = (size_t)(rowBase + r) * N + col;
    if (outF) outF[idx] = v;
    if (outB) outB[idx] = f2bf(v);
  }
}

__global__ __launch_bounds__(256) void gemm_bf16_wmma(
    const unsigned short* __restrict__ A,    // [M][K] bf16
    const unsigned short* __restrict__ Bt,   // [N][K] bf16 (B transposed)
    const float* __restrict__ bias,          // [N]
    float* __restrict__ outF,                // [M][N] fp32 or nullptr
    unsigned short* __restrict__ outB,       // [M][N] bf16 or nullptr
    int M, int N, int K)
{
  const int lane = threadIdx.x & 31;
  const int wave = threadIdx.x >> 5;
  const int wm = wave >> 2;                  // 0..1
  const int wn = wave & 3;                   // 0..3
  const int m0 = blockIdx.y * 64  + wm * 32;
  const int n0 = blockIdx.x * 128 + wn * 32;
  const int l16 = lane & 15;
  const int h   = lane >> 4;

  const unsigned short* __restrict__ pa0 = A  + (size_t)(m0 + l16)      * K;
  const unsigned short* __restrict__ pa1 = A  + (size_t)(m0 + 16 + l16) * K;
  const unsigned short* __restrict__ pb0 = Bt + (size_t)(n0 + l16)      * K;
  const unsigned short* __restrict__ pb1 = Bt + (size_t)(n0 + 16 + l16) * K;

  v8f c00 = {}, c01 = {}, c10 = {}, c11 = {};

#pragma unroll 2
  for (int kk = 0; kk < K; kk += 32) {
    if (kk + 128 < K) {                      // global_prefetch_b8 four steps ahead
      __builtin_prefetch(pa0 + kk + 128, 0, 0);
      __builtin_prefetch(pa1 + kk + 128, 0, 0);
      __builtin_prefetch(pb0 + kk + 128, 0, 0);
      __builtin_prefetch(pb1 + kk + 128, 0, 0);
    }
    // A fragment (ISA 16-bit 16x32 layout): lane half h owns K = h*8..+7 and
    // 16+h*8..+7  -> two b128 loads, 32B apart.
    FragBF a0, a1, b0, b1;
    {
      const v4u* p = (const v4u*)(pa0 + kk + h * 8);
      a0.u[0] = p[0]; a0.u[1] = p[2];
      p = (const v4u*)(pa1 + kk + h * 8);
      a1.u[0] = p[0]; a1.u[1] = p[2];
      // B fragment: lane half h owns K = h*8..+7 and 16+h*8..+7 of column n.
      p = (const v4u*)(pb0 + kk + h * 8);
      b0.u[0] = p[0]; b0.u[1] = p[2];
      p = (const v4u*)(pb1 + kk + h * 8);
      b1.u[0] = p[0]; b1.u[1] = p[2];
    }
    c00 = __builtin_amdgcn_wmma_f32_16x16x32_bf16(false, a0.v, false, b0.v, (short)0, c00, false, false);
    c01 = __builtin_amdgcn_wmma_f32_16x16x32_bf16(false, a0.v, false, b1.v, (short)0, c01, false, false);
    c10 = __builtin_amdgcn_wmma_f32_16x16x32_bf16(false, a1.v, false, b0.v, (short)0, c10, false, false);
    c11 = __builtin_amdgcn_wmma_f32_16x16x32_bf16(false, a1.v, false, b1.v, (short)0, c11, false, false);
  }

  const float bs0 = bias[n0 + l16];
  const float bs1 = bias[n0 + 16 + l16];
  // C/D layout: VGPR r holds M = r + 8*h (lanes 0-15 / 16-31), N = lane&15.
  store_tile(c00, m0 + 8 * h,      n0 + l16,      bs0, N, outF, outB);
  store_tile(c01, m0 + 8 * h,      n0 + 16 + l16, bs1, N, outF, outB);
  store_tile(c10, m0 + 16 + 8 * h, n0 + l16,      bs0, N, outF, outB);
  store_tile(c11, m0 + 16 + 8 * h, n0 + 16 + l16, bs1, N, outF, outB);
}

// ---------------- 4) passthrough / gather outputs ---------------------------
__global__ void gather_misc(const long long* __restrict__ sidx,
                            const long long* __restrict__ sdex,
                            const float* __restrict__ verts,
                            const float* __restrict__ j3d,
                            const float* __restrict__ j2d,
                            const long long* __restrict__ sides,
                            float* __restrict__ out)
{
  long long g = (long long)blockIdx.x * blockDim.x + threadIdx.x;
  if (g >= MISC_TOTAL) return;
  long long pos = (g < O_H) ? g : g + (O_BIDX - O_H);   // skip the h region
  float v;
  if (pos < O_DEX)        { int row = (int)pos;            v = (float)sidx[(row % 512) / 128]; }
  else if (pos < O_VERTS) { int row = (int)(pos - O_DEX);  v = (float)sdex[(row % 512) / 128]; }
  else if (pos < O_J3D)   { long long e = pos - O_VERTS; int row = (int)(e / 2334); int k = (int)(e % 2334);
                            v = verts[(size_t)((row % 512) / 128) * 2334 + k]; }
  else if (pos < O_J2D)   { long long e = pos - O_J3D;   int row = (int)(e / 63);   int k = (int)(e % 63);
                            v = j3d[((row % 512) / 128) * 63 + k]; }
  else if (pos < O_SIDES) { long long e = pos - O_J2D;   int row = (int)(e / 42);   int k = (int)(e % 42);
                            v = j2d[((row % 512) / 128) * 42 + k]; }
  else if (pos < O_H)     { int row = (int)(pos - O_SIDES); v = (float)sides[(row % 512) / 128]; }
  else if (pos < O_LIDX)  { int row = (int)(pos - O_BIDX);  v = (float)((row % 512) / 128); }
  else                    { int row = (int)(pos - O_LIDX);  v = (float)(row / 512); }
  out[pos] = v;
}

// ---------------- launcher --------------------------------------------------
extern "C" void kernel_launch(void* const* d_in, const int* in_sizes, int n_in,
                              void* d_out, int out_size, void* d_ws, size_t ws_size,
                              hipStream_t stream)
{
  const float*     f0    = (const float*)d_in[0];
  const float*     f1    = (const float*)d_in[1];
  const float*     f2    = (const float*)d_in[2];
  const float*     bx0   = (const float*)d_in[3];
  const float*     bx1   = (const float*)d_in[4];
  const float*     bx2   = (const float*)d_in[5];
  const long long* sidx  = (const long long*)d_in[6];
  const long long* sdex  = (const long long*)d_in[7];
  const float*     verts = (const float*)d_in[8];
  const float*     j3d   = (const float*)d_in[9];
  const float*     j2d   = (const float*)d_in[10];
  const long long* sides = (const long long*)d_in[11];
  const float*     w6    = (const float*)d_in[12];
  const float*     b6    = (const float*)d_in[13];
  const float*     w7    = (const float*)d_in[14];
  const float*     b7    = (const float*)d_in[15];

  // workspace layout (bytes, all 16B aligned):
  //   A1  bf16 [1536][12544] : 38,535,168
  //   W6t bf16 [1024][12544] : 25,690,112
  //   H1  bf16 [1536][1024]  :  3,145,728
  //   W7t bf16 [1024][1024]  :  2,097,152   (total ~69.5 MB)
  char* ws = (char*)d_ws;
  unsigned short* A1  = (unsigned short*)(ws);
  unsigned short* W6t = (unsigned short*)(ws + 38535168);
  unsigned short* H1  = (unsigned short*)(ws + 64225280);
  unsigned short* W7t = (unsigned short*)(ws + 67371008);
  float* outH = (float*)d_out + O_H;

  roi_align_bf16<<<NROI, 256, 0, stream>>>(f0, f1, f2, bx0, bx1, bx2, A1);
  transpose_cvt_bf16<<<dim3(REP / 32, FLATK / 32), dim3(32, 8), 0, stream>>>(w6, W6t, FLATK, REP);
  transpose_cvt_bf16<<<dim3(REP / 32, REP / 32),   dim3(32, 8), 0, stream>>>(w7, W7t, REP, REP);

  // GEMM1: h1 = relu(A1 @ w6 + b6), bf16 out to ws
  gemm_bf16_wmma<<<dim3(REP / 128, NROI / 64), 256, 0, stream>>>(
      A1, W6t, b6, nullptr, H1, NROI, REP, FLATK);
  // GEMM2: h = relu(h1 @ w7 + b7), fp32 out straight into d_out
  gemm_bf16_wmma<<<dim3(REP / 128, NROI / 64), 256, 0, stream>>>(
      H1, W7t, b7, outH, nullptr, NROI, REP, REP);

  gather_misc<<<(int)((MISC_TOTAL + 255) / 256), 256, 0, stream>>>(
      sidx, sdex, verts, j3d, j2d, sides, (float*)d_out);
}
